// MoE_MLP_19387482374735
// MI455X (gfx1250) — matrix operand order repaired and checked by reading, not compiled, not run on hardware
//
#include <hip/hip_runtime.h>

#define NTOK 65536
#define DIN  128
#define DH   256
#define DOUT 128
#define NE   8

typedef __attribute__((ext_vector_type(16))) __bf16 v16bf;
typedef __attribute__((ext_vector_type(8)))  float  v8f;

union BFrag { v16bf v; unsigned short u[16]; uint4 q[2]; };

__device__ inline unsigned short f2bf(float f) {
  unsigned int u = __float_as_uint(f);
  unsigned int r = u + 0x7FFFu + ((u >> 16) & 1u);   // round-to-nearest-even
  return (unsigned short)(r >> 16);
}

__device__ inline v8f wmma_bf16(v16bf a, v16bf b, v8f c) {
  return __builtin_amdgcn_wmma_f32_16x16x32_bf16(false, a, false, b, (short)0, c, false, false);
}

// ---------------- zero output + expert counters ----------------
__global__ void zero_out(float* __restrict__ out, int* __restrict__ cnt, int n) {
  int i = blockIdx.x * blockDim.x + threadIdx.x;
  if (i < n)  out[i] = 0.0f;
  if (i < NE) cnt[i] = 0;
}

// ---------- fp32 [b][K][Nc] -> bf16 transposed [b][Nc][K] ----------
__global__ void cvt_transpose(const float* __restrict__ in, unsigned short* __restrict__ outw,
                              int K, int Nc, int total) {
  int i = blockIdx.x * blockDim.x + threadIdx.x;
  if (i >= total) return;
  int k = i % K;
  int n = (i / K) % Nc;
  int b = i / (K * Nc);
  outw[i] = f2bf(in[(size_t)b * K * Nc + (size_t)k * Nc + n]);
}

// ---------------- router: gate, top-2, softmax, bin ----------------
__global__ __launch_bounds__(256) void router(const float* __restrict__ x, const float* __restrict__ Wg,
                                              int* __restrict__ cnt, int* __restrict__ tokl,
                                              float* __restrict__ wtsl) {
  __shared__ float wg[DIN * NE];
  for (int i = threadIdx.x; i < DIN * NE; i += 256) wg[i] = Wg[i];
  __syncthreads();
  int n = blockIdx.x * 256 + threadIdx.x;
  const float* xr = x + (size_t)n * DIN;
  float acc[NE];
#pragma unroll
  for (int e = 0; e < NE; ++e) acc[e] = 0.0f;
  for (int d = 0; d < DIN; ++d) {
    float xv = xr[d];
#pragma unroll
    for (int e = 0; e < NE; ++e) acc[e] += xv * wg[d * NE + e];
  }
  int b0 = 0; float v0 = acc[0];
#pragma unroll
  for (int e = 1; e < NE; ++e) if (acc[e] > v0) { v0 = acc[e]; b0 = e; }
  int b1 = (b0 == 0) ? 1 : 0; float v1 = acc[b1];
#pragma unroll
  for (int e = 0; e < NE; ++e) if (e != b0 && acc[e] > v1) { v1 = acc[e]; b1 = e; }
  float t  = __expf(v1 - v0);          // softmax over the 2 selected logits
  float w0 = 1.0f / (1.0f + t);
  float w1 = t * w0;
  int p0 = atomicAdd(&cnt[b0], 1);
  tokl[b0 * NTOK + p0] = n; wtsl[b0 * NTOK + p0] = w0;
  int p1 = atomicAdd(&cnt[b1], 1);
  tokl[b1 * NTOK + p1] = n; wtsl[b1 * NTOK + p1] = w1;
}

// ------- main expert MLP: 1 wave / workgroup, M=32 tokens / wave -------
// Each B fragment (32B/lane, one n-tile x 32-K chunk of W^T) feeds TWO
// WMMAs (two 16-row A tiles) -> 1 global_load_b128 per v_wmma.
__global__ __launch_bounds__(32) void moe_main(
    const float* __restrict__ x,
    const float* __restrict__ b1, const float* __restrict__ b2, const float* __restrict__ bo,
    const unsigned short* __restrict__ W1t, const unsigned short* __restrict__ W2t,
    const unsigned short* __restrict__ Wot,
    const int* __restrict__ cnt, const int* __restrict__ tokl, const float* __restrict__ wtsl,
    float* __restrict__ out)
{
  int e = blockIdx.x;
  int count = cnt[e];
  int tile0 = blockIdx.y * 32;
  if (tile0 >= count) return;                 // block-uniform

  __shared__ unsigned short h1[32][DH];       // 32x256 bf16 activations
  __shared__ unsigned short h2[32][DH];
  __shared__ int   stok[32];
  __shared__ float swt[32];

  int lane = threadIdx.x;                     // 0..31, single wave
  int nn   = lane & 15;                       // column / row-within-half
  int kh   = lane >> 4;                       // lane-group half

  {
    int idx = tile0 + lane;
    int t = 0; float w = 0.0f;                // pad rows: weight 0 -> no contribution
    if (idx < count) { t = tokl[e * NTOK + idx]; w = wtsl[e * NTOK + idx]; }
    stok[lane] = t; swt[lane] = w;
  }
  __syncthreads();                            // single-wave WG: S_NOP + LDS fence

  // ---- A1 fragments for both 16-row tiles: fp32 x -> bf16, ISA A layout ----
  v16bf a1[2][4];
#pragma unroll
  for (int t = 0; t < 2; ++t) {
    const float* xr = x + (size_t)stok[t * 16 + nn] * DIN;
#pragma unroll
    for (int k = 0; k < 4; ++k) {
      int kb = k * 32 + kh * 8;
      float4 c0 = *(const float4*)(xr + kb);
      float4 c1 = *(const float4*)(xr + kb + 4);
      float4 c2 = *(const float4*)(xr + kb + 16);
      float4 c3 = *(const float4*)(xr + kb + 20);
      BFrag f;
      f.u[0]  = f2bf(c0.x); f.u[1]  = f2bf(c0.y); f.u[2]  = f2bf(c0.z); f.u[3]  = f2bf(c0.w);
      f.u[4]  = f2bf(c1.x); f.u[5]  = f2bf(c1.y); f.u[6]  = f2bf(c1.z); f.u[7]  = f2bf(c1.w);
      f.u[8]  = f2bf(c2.x); f.u[9]  = f2bf(c2.y); f.u[10] = f2bf(c2.z); f.u[11] = f2bf(c2.w);
      f.u[12] = f2bf(c3.x); f.u[13] = f2bf(c3.y); f.u[14] = f2bf(c3.z); f.u[15] = f2bf(c3.w);
      a1[t][k] = f.v;
    }
  }

  // ---- phase 1: h1 = relu(x @ W1 + b1), K=128 ----
  const unsigned short* W1e = W1t + (size_t)e * DH * DIN;
  for (int n0 = 0; n0 < DH; n0 += 16) {
    float bias = b1[e * DH + n0 + nn];
    v8f acc0, acc1;
#pragma unroll
    for (int r = 0; r < 8; ++r) { acc0[r] = bias; acc1[r] = bias; }
    const unsigned short* bp = W1e + (size_t)(n0 + nn) * DIN + kh * 8;
#pragma unroll
    for (int k = 0; k < 4; ++k) {
      BFrag bf;
      bf.q[0] = *(const uint4*)(bp + k * 32);
      bf.q[1] = *(const uint4*)(bp + k * 32 + 16);
      acc0 = wmma_bf16(a1[0][k], bf.v, acc0);
      acc1 = wmma_bf16(a1[1][k], bf.v, acc1);
    }
#pragma unroll
    for (int r = 0; r < 8; ++r) {
      h1[r + 8 * kh][n0 + nn]      = f2bf(fmaxf(acc0[r], 0.0f));
      h1[16 + r + 8 * kh][n0 + nn] = f2bf(fmaxf(acc1[r], 0.0f));
    }
  }
  __syncthreads();

  // ---- phase 2: h2 = relu(h1 @ W2 + b2), K=256 ----
  v16bf a2[2][8];
#pragma unroll
  for (int t = 0; t < 2; ++t)
#pragma unroll
    for (int k = 0; k < 8; ++k) {
      const unsigned short* p = &h1[t * 16 + nn][k * 32 + kh * 8];
      BFrag f;
      f.q[0] = *(const uint4*)p;
      f.q[1] = *(const uint4*)(p + 16);
      a2[t][k] = f.v;
    }
  const unsigned short* W2e = W2t + (size_t)e * DH * DH;
  for (int n0 = 0; n0 < DH; n0 += 16) {
    float bias = b2[e * DH + n0 + nn];
    v8f acc0, acc1;
#pragma unroll
    for (int r = 0; r < 8; ++r) { acc0[r] = bias; acc1[r] = bias; }
    const unsigned short* bp = W2e + (size_t)(n0 + nn) * DH + kh * 8;
#pragma unroll
    for (int k = 0; k < 8; ++k) {
      BFrag bf;
      bf.q[0] = *(const uint4*)(bp + k * 32);
      bf.q[1] = *(const uint4*)(bp + k * 32 + 16);
      acc0 = wmma_bf16(a2[0][k], bf.v, acc0);
      acc1 = wmma_bf16(a2[1][k], bf.v, acc1);
    }
#pragma unroll
    for (int r = 0; r < 8; ++r) {
      h2[r + 8 * kh][n0 + nn]      = f2bf(fmaxf(acc0[r], 0.0f));
      h2[16 + r + 8 * kh][n0 + nn] = f2bf(fmaxf(acc1[r], 0.0f));
    }
  }
  __syncthreads();

  // ---- phase 3: y = h2 @ Wo + bo; out += w * y ----
  v16bf a3[2][8];
#pragma unroll
  for (int t = 0; t < 2; ++t)
#pragma unroll
    for (int k = 0; k < 8; ++k) {
      const unsigned short* p = &h2[t * 16 + nn][k * 32 + kh * 8];
      BFrag f;
      f.q[0] = *(const uint4*)p;
      f.q[1] = *(const uint4*)(p + 16);
      a3[t][k] = f.v;
    }
  float wv0[8], wv1[8]; int tv0[8], tv1[8];
#pragma unroll
  for (int r = 0; r < 8; ++r) {
    wv0[r] = swt[r + 8 * kh];      tv0[r] = stok[r + 8 * kh];
    wv1[r] = swt[16 + r + 8 * kh]; tv1[r] = stok[16 + r + 8 * kh];
  }
  for (int n0 = 0; n0 < DOUT; n0 += 16) {
    float bias = bo[n0 + nn];
    v8f acc0, acc1;
#pragma unroll
    for (int r = 0; r < 8; ++r) { acc0[r] = bias; acc1[r] = bias; }
    const unsigned short* bp = Wot + (size_t)(n0 + nn) * DH + kh * 8;
#pragma unroll
    for (int k = 0; k < 8; ++k) {
      BFrag bf;
      bf.q[0] = *(const uint4*)(bp + k * 32);
      bf.q[1] = *(const uint4*)(bp + k * 32 + 16);
      acc0 = wmma_bf16(a3[0][k], bf.v, acc0);
      acc1 = wmma_bf16(a3[1][k], bf.v, acc1);
    }
#pragma unroll
    for (int r = 0; r < 8; ++r) {
      // exactly 2 commutative f32 contributions per output element -> deterministic
      __hip_atomic_fetch_add(out + (size_t)tv0[r] * DOUT + n0 + nn, acc0[r] * wv0[r],
                             __ATOMIC_RELAXED, __HIP_MEMORY_SCOPE_AGENT);
      __hip_atomic_fetch_add(out + (size_t)tv1[r] * DOUT + n0 + nn, acc1[r] * wv1[r],
                             __ATOMIC_RELAXED, __HIP_MEMORY_SCOPE_AGENT);
    }
  }
}

extern "C" void kernel_launch(void* const* d_in, const int* in_sizes, int n_in,
                              void* d_out, int out_size, void* d_ws, size_t ws_size,
                              hipStream_t stream) {
  const float* x  = (const float*)d_in[0];
  const float* Wg = (const float*)d_in[1];
  const float* W1 = (const float*)d_in[2];
  const float* b1 = (const float*)d_in[3];
  const float* W2 = (const float*)d_in[4];
  const float* b2 = (const float*)d_in[5];
  const float* Wo = (const float*)d_in[6];
  const float* bo = (const float*)d_in[7];
  float* out = (float*)d_out;

  char* ws = (char*)d_ws;
  size_t off = 0;
  int*   cnt  = (int*)(ws + off);  off += 256;
  int*   tokl = (int*)(ws + off);  off += (size_t)NE * NTOK * sizeof(int);
  float* wtsl = (float*)(ws + off); off += (size_t)NE * NTOK * sizeof(float);
  unsigned short* W1t = (unsigned short*)(ws + off); off += (size_t)NE * DH * DIN * 2;
  unsigned short* W2t = (unsigned short*)(ws + off); off += (size_t)NE * DH * DH * 2;
  unsigned short* Wot = (unsigned short*)(ws + off); off += (size_t)DOUT * DH * 2;

  int totOut = NTOK * DOUT;
  zero_out<<<(totOut + 255) / 256, 256, 0, stream>>>(out, cnt, totOut);

  cvt_transpose<<<(NE * DIN * DH + 255) / 256, 256, 0, stream>>>(W1, W1t, DIN, DH, NE * DIN * DH);
  cvt_transpose<<<(NE * DH * DH + 255) / 256, 256, 0, stream>>>(W2, W2t, DH, DH, NE * DH * DH);
  cvt_transpose<<<(DH * DOUT + 255) / 256, 256, 0, stream>>>(Wo, Wot, DH, DOUT, DH * DOUT);

  router<<<NTOK / 256, 256, 0, stream>>>(x, Wg, cnt, tokl, wtsl);

  dim3 g(NE, NTOK / 32);   // capacity grid; blocks past cnt[e] early-exit
  moe_main<<<g, 32, 0, stream>>>(x, b1, b2, bo, W1t, W2t, Wot, cnt, tokl, wtsl, out);
}
